// EnhancedROIPool_81784767250595
// MI455X (gfx1250) — matrix operand order repaired and checked by reading, not compiled, not run on hardware
//
#include <hip/hip_runtime.h>
#include <hip/hip_bf16.h>
#include <stdint.h>

// ---------------- problem constants (fixed by the reference) ----------------
#define N_NODES   131072
#define B_GRAPHS  512
#define F_IN      512          // in_channels
#define H_DIM     256          // hidden = F/2
#define NPG       256          // nodes per graph = N/B
#define KSEL      128          // max(1, floor(0.5 * 256))
#define MARGIN_C  0.5f
#define TOPK_W    0.2f

// output layout: [pooled 512*512][loss 1][mask 131072]
#define POOL_OFF  0
#define LOSS_OFF  (B_GRAPHS * F_IN)
#define MASK_OFF  (LOSS_OFF + 1)

#define ROWS_PER_BLK 64        // 4 row-tiles of 16
#define CT_PER_WAVE  4         // hidden col-tiles per wave

typedef __attribute__((ext_vector_type(16))) __bf16 v16bf;
typedef __attribute__((ext_vector_type(8)))  float  v8f;

__device__ __forceinline__ unsigned short f2bf(float f) {
    union { float f; unsigned u; } c; c.f = f;
    unsigned u = c.u;
    u += 0x7FFFu + ((u >> 16) & 1u);        // round-to-nearest-even
    return (unsigned short)(u >> 16);
}

// XOR-mode ds_swizzle lane reduction step (stays within 16-lane halves)
__device__ __forceinline__ float swz_xor(float v, int imm) {
    int r;
    switch (imm) {   // offset = (xor_mask << 10) | and_mask(0x1F)
        case 1:  r = __builtin_amdgcn_ds_swizzle(__builtin_bit_cast(int, v), 0x041F); break;
        case 2:  r = __builtin_amdgcn_ds_swizzle(__builtin_bit_cast(int, v), 0x081F); break;
        case 4:  r = __builtin_amdgcn_ds_swizzle(__builtin_bit_cast(int, v), 0x101F); break;
        default: r = __builtin_amdgcn_ds_swizzle(__builtin_bit_cast(int, v), 0x201F); break;
    }
    return __builtin_bit_cast(float, r);
}

// ---------------------------------------------------------------------------
// Kernel 0: repack W1 [512 x 256] f32 (row-major) into bf16 WMMA B-fragments.
// This code's contiguous K order (16-bit B, 32x16): lane l holds column l&15; lanes 0-15 K=0..15
// (VGPR v: K=2v,2v+1), lanes 16-31 K=16..31.
// ws layout: Bp[((tile*16 + kchunk)*32 + lane)*8 + v]  (uint = 2 packed bf16)
// ---------------------------------------------------------------------------
__global__ void repack_w1(const float* __restrict__ W1, unsigned* __restrict__ Bp) {
    int gid = blockIdx.x * blockDim.x + threadIdx.x;   // 0..8191
    int l   = gid & 31;
    int kc  = (gid >> 5) & 15;                         // K chunk (32 each)
    int t   = gid >> 9;                                // hidden-col tile (16 each)
    int col = t * 16 + (l & 15);
    int kh  = (l >> 4) * 16;
    unsigned* dst = Bp + (size_t)gid * 8;
#pragma unroll
    for (int v = 0; v < 8; ++v) {
        int k = kc * 32 + kh + 2 * v;
        unsigned short lo = f2bf(W1[(size_t)k       * H_DIM + col]);
        unsigned short hi = f2bf(W1[(size_t)(k + 1) * H_DIM + col]);
        dst[v] = (unsigned)lo | ((unsigned)hi << 16);
    }
}

// ---------------------------------------------------------------------------
// Kernel 1: fused score-net. Block = 64 rows, 512 threads = 16 waves.
// Wave w: row-tile rt = w&3, col-group cg = w>>2 (4 col-tiles of 16).
// Each A fragment (one ds-load pair per kc) feeds 4 WMMAs -> 4x less DS
// traffic per row than a 1-acc scheme. Epilogue fuses +b1, ReLU, *W2 and a
// ds_swizzle lane reduction into per-row score partials.
// ---------------------------------------------------------------------------
__global__ __launch_bounds__(512)
void score_net(const float* __restrict__ x, const unsigned* __restrict__ Bp,
               const float* __restrict__ b1, const float* __restrict__ W2,
               const float* __restrict__ b2, float* __restrict__ scores) {
    __shared__ __align__(16) unsigned short ldsx[ROWS_PER_BLK * F_IN];  // 64 KB
    __shared__ float part[16][16];                                      // [wave][rowInTile]

    const int tid     = threadIdx.x;
    const int wv      = tid >> 5;
    const int lane    = tid & 31;
    const int rowbase = blockIdx.x * ROWS_PER_BLK;

    // Stage 64x512 f32 -> bf16 into LDS (coalesced float2 loads, packed stores)
    {
        const float2* xs = (const float2*)(x + (size_t)rowbase * F_IN);
        unsigned* lx = (unsigned*)ldsx;
#pragma unroll 8
        for (int i = 0; i < ROWS_PER_BLK * F_IN / 2 / 512; ++i) {   // 32 iters
            int idx = tid + i * 512;
            float2 v = xs[idx];
            lx[idx] = (unsigned)f2bf(v.x) | ((unsigned)f2bf(v.y) << 16);
        }
    }
    __syncthreads();

    const int half = lane >> 4;               // A-frag K-half select
    const int m    = lane & 15;               // A row / B,C column
    const int rt   = wv & 3;                  // row-tile within block
    const int cg   = wv >> 2;                 // col-group (4 tiles of 16)

    v8f c[CT_PER_WAVE];
#pragma unroll
    for (int ct = 0; ct < CT_PER_WAVE; ++ct) c[ct] = (v8f){0.f,0.f,0.f,0.f,0.f,0.f,0.f,0.f};

#pragma unroll 2
    for (int kc = 0; kc < 16; ++kc) {
        union { v16bf v; uint4 q[2]; } a;
        // A (16-bit 16x32 layout): lanes 0-15 -> K {0..15},
        // lanes 16-31 -> K {16..31}; 16 contiguous bf16 each (matches B).
        const int kb = kc * 32 + half * 16;
        const int arow = (rt * 16 + m) * F_IN;
        a.q[0] = *(const uint4*)&ldsx[arow + kb];
        a.q[1] = *(const uint4*)&ldsx[arow + kb + 8];
#pragma unroll
        for (int ct = 0; ct < CT_PER_WAVE; ++ct) {
            union { v16bf v; uint4 q[2]; } b;
            const int tile = cg * CT_PER_WAVE + ct;
            const uint4* bp = (const uint4*)(Bp + (((size_t)(tile * 16 + kc) * 32) + lane) * 8);
            b.q[0] = bp[0];
            b.q[1] = bp[1];
            c[ct] = __builtin_amdgcn_wmma_f32_16x16x32_bf16(false, a.v, false, b.v,
                                                            (short)0, c[ct], false, false);
        }
    }

    // hidden = relu(c + b1[col]); score partial = sum_col hidden * W2[col]
    float sr[8] = {0.f,0.f,0.f,0.f,0.f,0.f,0.f,0.f};
#pragma unroll
    for (int ct = 0; ct < CT_PER_WAVE; ++ct) {
        const int col = cg * (CT_PER_WAVE * 16) + ct * 16 + m;
        const float w2v = W2[col];
        const float b1v = b1[col];
#pragma unroll
        for (int r = 0; r < 8; ++r) {
            float h = c[ct][r] + b1v;
            h = fmaxf(h, 0.f);
            sr[r] += h * w2v;                 // row M = r + 8*half (in tile rt)
        }
    }
    // reduce over the 16 columns (lanes within each 16-lane half)
#pragma unroll
    for (int r = 0; r < 8; ++r) {
        sr[r] += swz_xor(sr[r], 1);
        sr[r] += swz_xor(sr[r], 2);
        sr[r] += swz_xor(sr[r], 4);
        sr[r] += swz_xor(sr[r], 8);
    }
    if (m == 0) {
        const int rb = half * 8;
#pragma unroll
        for (int r = 0; r < 8; ++r) part[wv][rb + r] = sr[r];
    }
    __syncthreads();

    if (tid < ROWS_PER_BLK) {                 // reduce over the 4 col-groups
        const int rti = tid >> 4;
        const int rr  = tid & 15;
        float s = b2[0] + part[rti][rr] + part[rti + 4][rr]
                        + part[rti + 8][rr] + part[rti + 12][rr];
        scores[rowbase + tid] = s;
    }
}

// ---------------------------------------------------------------------------
// Kernel 2: per-graph top-k select (stable rank-by-count, matches lexsort),
// loss stats (deterministic tree reduce), and masked pooling with a
// block-uniform skip branch (reads only selected rows of X, float2 vector).
// ---------------------------------------------------------------------------
__global__ __launch_bounds__(256)
void select_pool(const float* __restrict__ x, const float* __restrict__ scores,
                 float* __restrict__ out_pool, float* __restrict__ out_mask,
                 float* __restrict__ loss_pg) {
    __shared__ float s[NPG];
    __shared__ float mk[NPG];
    __shared__ float red[NPG];
    __shared__ float tot_s;

    const int g = blockIdx.x;
    const int t = threadIdx.x;
    const int base = g * NPG;

    const float ms = scores[base + t];
    s[t] = ms;
    __syncthreads();

    // stable descending rank: # of j with (s_j > s_t) or tie with smaller index
    int cnt = 0;
#pragma unroll 8
    for (int j = 0; j < NPG; ++j) {
        float sj = s[j];
        cnt += (sj > ms) || (sj == ms && j < t);
    }
    const float sel = (cnt < KSEL) ? 1.f : 0.f;
    mk[t] = sel;
    out_mask[base + t] = sel;

    // tot_sum
    red[t] = ms;
    __syncthreads();
    for (int o = NPG / 2; o > 0; o >>= 1) {
        if (t < o) red[t] += red[t + o];
        __syncthreads();
    }
    if (t == 0) tot_s = red[0];
    __syncthreads();
    // sel_sum
    red[t] = ms * sel;
    __syncthreads();
    for (int o = NPG / 2; o > 0; o >>= 1) {
        if (t < o) red[t] += red[t + o];
        __syncthreads();
    }
    if (t == 0) {
        const float sel_sum  = red[0];
        const float sel_mean = sel_sum / (float)KSEL;
        const float uns_mean = (tot_s - sel_sum) / (float)(NPG - KSEL);
        loss_pg[g] = fmaxf(MARGIN_C - (sel_mean - uns_mean), 0.f);
    }

    // masked pooling: thread t owns float2 at cols [2t, 2t+1]; uniform branch
    float2 acc = {0.f, 0.f};
    const float* xg = x + (size_t)base * F_IN;
#pragma unroll 4
    for (int n = 0; n < NPG; ++n) {
        if (mk[n] != 0.f) {
            float2 v = ((const float2*)(xg + (size_t)n * F_IN))[t];
            acc.x += v.x;
            acc.y += v.y;
        }
    }
    ((float2*)(out_pool + (size_t)g * F_IN))[t] = acc;
}

// ---------------------------------------------------------------------------
// Kernel 3: deterministic fixed-order loss reduction (no float atomics).
// ---------------------------------------------------------------------------
__global__ void loss_reduce(const float* __restrict__ loss_pg, float* __restrict__ out_loss) {
    __shared__ float red[B_GRAPHS];
    const int t = threadIdx.x;
    red[t] = loss_pg[t];
    __syncthreads();
    for (int o = B_GRAPHS / 2; o > 0; o >>= 1) {
        if (t < o) red[t] += red[t + o];
        __syncthreads();
    }
    if (t == 0) out_loss[0] = red[0] * (TOPK_W / (float)B_GRAPHS);
}

// ---------------------------------------------------------------------------
extern "C" void kernel_launch(void* const* d_in, const int* in_sizes, int n_in,
                              void* d_out, int out_size, void* d_ws, size_t ws_size,
                              hipStream_t stream) {
    const float* x  = (const float*)d_in[0];
    // d_in[1] = batch ids (int32): structure is fixed (equal segments), unused
    const float* W1 = (const float*)d_in[2];
    const float* b1 = (const float*)d_in[3];
    const float* W2 = (const float*)d_in[4];
    const float* b2 = (const float*)d_in[5];

    float* out      = (float*)d_out;
    float* out_pool = out + POOL_OFF;
    float* out_loss = out + LOSS_OFF;
    float* out_mask = out + MASK_OFF;

    // workspace: [Bp: 65536 uints = 256KB][scores: N floats][loss_pg: B floats]
    unsigned* Bp     = (unsigned*)d_ws;
    float*    scores = (float*)d_ws + 65536;
    float*    lpg    = scores + N_NODES;

    hipLaunchKernelGGL(repack_w1, dim3(32), dim3(256), 0, stream, W1, Bp);
    hipLaunchKernelGGL(score_net, dim3(N_NODES / ROWS_PER_BLK), dim3(512), 0, stream,
                       x, Bp, b1, W2, b2, scores);
    hipLaunchKernelGGL(select_pool, dim3(B_GRAPHS), dim3(256), 0, stream,
                       x, scores, out_pool, out_mask, lpg);
    hipLaunchKernelGGL(loss_reduce, dim3(1), dim3(B_GRAPHS), 0, stream,
                       lpg, out_loss);
}
